// MambaMixer_3882650436909
// MI455X (gfx1250) — matrix operand order repaired and checked by reading, not compiled, not run on hardware
//
#include <hip/hip_runtime.h>
#include <hip/hip_bf16.h>
#include <math.h>

// ---------------------------------------------------------------------------
// Mamba mixer forward for MI455X (gfx1250, wave32).
// B=2, L=2048, DM=1024, DIN=2048, DST=16, DCONV=4, RDT=64.
// GEMMs: bf16 operands staged into LDS with GLOBAL_LOAD_ASYNC_TO_LDS_B128
// (ASYNCcnt), consumed by v_wmma_f32_16x16x32_bf16 (f32 accumulate).
// Selective scan: one state element per lane + shfl reductions.
// ---------------------------------------------------------------------------

#define BSZ   2
#define LSEQ  2048
#define DM    1024
#define DST   16
#define DCONV 4
#define DIN   2048
#define RDT   64
#define NROW  (BSZ * LSEQ)        // 4096
#define XPROJ_N 96
#define XPROJ_PAD 128             // zero-padded rows so staging is guard-free

typedef __attribute__((ext_vector_type(16))) __bf16 v16bf;
typedef __attribute__((ext_vector_type(8)))  float  v8f;

// ---------------------------------------------------------------------------
// GEMM: C[M,N] = A[M,K](bf16) * W[N,K](bf16)^T. Block tile 64(M) x 128(N),
// 256 threads = 8 waves; each wave: 1 A fragment reused across 4 WMMAs.
// K % 32 == 0, M % 64 == 0. W must be padded so nbase+127 rows are readable;
// C stores are N-guarded. store_bf16 selects output element type.
// Staging: one async b128 per lane for A (64x32), two for W (128x32).
// ---------------------------------------------------------------------------
__global__ __launch_bounds__(256)
void mamba_wmma_gemm(const __bf16* __restrict__ A, const __bf16* __restrict__ W,
                     void* __restrict__ Cout, int N, int K, int lda, int ldc,
                     int store_bf16)
{
    __shared__ __bf16 As[64 * 32];    // 4 KB
    __shared__ __bf16 Bs[128 * 32];   // 8 KB

    const int tid  = threadIdx.x;
    const int lane = tid & 31;
    const int wave = tid >> 5;       // 0..7
    const int wrow = wave & 3;       // M subtile (0..3)
    const int wcol = wave >> 2;      // N half (0..1), 64 cols each

    const int mbase = blockIdx.x * 64;
    const int nbase = blockIdx.y * 128;

    // per-lane staging chunk: 16B = 8 bf16
    const int srow = tid >> 2;          // 0..63
    const int scol = (tid & 3) * 8;     // 0,8,16,24

    const unsigned ldsA  = (unsigned)(size_t)(As + srow * 32 + scol);
    const unsigned ldsB0 = (unsigned)(size_t)(Bs + srow * 32 + scol);
    const unsigned ldsB1 = (unsigned)(size_t)(Bs + (64 + srow) * 32 + scol);

    const __bf16* gA  = A + (size_t)(mbase + srow) * lda + scol;
    const __bf16* gB0 = W + (size_t)(nbase + srow) * K + scol;
    const __bf16* gB1 = W + (size_t)(nbase + 64 + srow) * K + scol;

    v8f c0 = {}, c1 = {}, c2 = {}, c3 = {};

    const int r    = lane & 15;
    const int half = lane >> 4;

    for (int k0 = 0; k0 < K; k0 += 32) {
        // ---- async global -> LDS staging (16B per lane per tile-slice) ----
        asm volatile("global_load_async_to_lds_b128 %0, %1, off"
                     :: "v"(ldsA),  "v"(gA  + k0) : "memory");
        asm volatile("global_load_async_to_lds_b128 %0, %1, off"
                     :: "v"(ldsB0), "v"(gB0 + k0) : "memory");
        asm volatile("global_load_async_to_lds_b128 %0, %1, off"
                     :: "v"(ldsB1), "v"(gB1 + k0) : "memory");
        asm volatile("s_wait_asynccnt 0" ::: "memory");
        __syncthreads();

        // ---- A fragment: lane holds row M=r; K = [8h..8h+7] and [16+8h..] ----
        v16bf a;
        {
            const __bf16* arow = As + (16 * wrow + r) * 32;
            #pragma unroll
            for (int i = 0; i < 8; ++i) {
                a[i]     = arow[8 * half + i];
                a[8 + i] = arow[16 + 8 * half + i];
            }
        }
        // ---- 4 x (B fragment + WMMA); lane holds col N=r; K = [8h..8h+7] and [16+8h..] ----
        #pragma unroll
        for (int s = 0; s < 4; ++s) {
            v16bf b;
            const __bf16* bp = Bs + (64 * wcol + 16 * s + r) * 32;
            #pragma unroll
            for (int i = 0; i < 8; ++i) {
                b[i]     = bp[8 * half + i];
                b[8 + i] = bp[16 + 8 * half + i];
            }
            v8f& acc = (s == 0) ? c0 : (s == 1) ? c1 : (s == 2) ? c2 : c3;
            acc = __builtin_amdgcn_wmma_f32_16x16x32_bf16(false, a, false, b,
                                                          (short)0, acc,
                                                          false, false);
        }
        __syncthreads();
    }

    // ---- store: VGPR j -> M = j / 8+j by lane half, N = lane&15 ----
    const int mrow = mbase + 16 * wrow + half * 8;
    #pragma unroll
    for (int s = 0; s < 4; ++s) {
        const v8f& acc = (s == 0) ? c0 : (s == 1) ? c1 : (s == 2) ? c2 : c3;
        const int nc = nbase + 64 * wcol + 16 * s + r;
        if (nc < N) {
            if (store_bf16) {
                __bf16* Cb = (__bf16*)Cout;
                #pragma unroll
                for (int j = 0; j < 8; ++j)
                    Cb[(size_t)(mrow + j) * ldc + nc] = (__bf16)acc[j];
            } else {
                float* Cf = (float*)Cout;
                #pragma unroll
                for (int j = 0; j < 8; ++j)
                    Cf[(size_t)(mrow + j) * ldc + nc] = acc[j];
            }
        }
    }
}

// ---------------------------------------------------------------------------
// fp32 -> bf16 elementwise convert.
// ---------------------------------------------------------------------------
__global__ __launch_bounds__(256)
void cvt_f32_bf16(const float* __restrict__ src, __bf16* __restrict__ dst)
{
    const int i = blockIdx.x * blockDim.x + threadIdx.x;
    dst[i] = (__bf16)src[i];
}

// fp32 (rows_src x cols) -> bf16 (rows_dst x cols) with zero row padding.
__global__ __launch_bounds__(256)
void cvt_pad_bf16(const float* __restrict__ src, __bf16* __restrict__ dst,
                  int rows_src, int cols)
{
    const int i = blockIdx.x * blockDim.x + threadIdx.x;
    const int r = i / cols;
    dst[i] = (r < rows_src) ? (__bf16)src[i] : (__bf16)0.0f;
}

// extract dt_lo slice of x_dbl (fp32, row stride 96, cols 0..63) -> bf16
__global__ __launch_bounds__(256)
void extract_dtlo(const float* __restrict__ xdbl, __bf16* __restrict__ dst)
{
    const int i = blockIdx.x * blockDim.x + threadIdx.x;  // over NROW*RDT
    const int row = i >> 6;
    const int c   = i & 63;
    dst[i] = (__bf16)xdbl[(size_t)row * XPROJ_N + c];
}

// ---------------------------------------------------------------------------
// Depthwise causal conv1d (k=4, left pad 3) + bias + SiLU.
// xz is bf16 (B*L, 2*DIN); xs = cols [0,DIN). Outputs xc fp32 and bf16.
// ---------------------------------------------------------------------------
__global__ __launch_bounds__(256)
void mamba_conv_silu(const __bf16* __restrict__ xz, const float* __restrict__ Wc,
                     const float* __restrict__ bc, float* __restrict__ xc,
                     __bf16* __restrict__ xc_bf)
{
    const int idx = blockIdx.x * blockDim.x + threadIdx.x;  // over B*L*DIN
    const int d   = idx % DIN;
    const int bl  = idx / DIN;
    const int l   = bl % LSEQ;

    float acc = bc[d];
    #pragma unroll
    for (int k = 0; k < DCONV; ++k) {
        const int ll = l + k - (DCONV - 1);
        if (ll >= 0) {
            acc = fmaf((float)xz[(size_t)(bl + k - (DCONV - 1)) * (2 * DIN) + d],
                       Wc[d * DCONV + k], acc);
        }
    }
    const float s = acc / (1.0f + __expf(-acc));   // SiLU
    xc[(size_t)bl * DIN + d]    = s;
    xc_bf[(size_t)bl * DIN + d] = (__bf16)s;
}

// ---------------------------------------------------------------------------
// Selective scan. One lane per (b, d, n) state element: 65536 lanes.
// delta = softplus(dt + 2*b_dt) (reference adds the bias twice).
// h <- exp(delta*A)*h + delta*Bv[n]*u ; y = sum_n h*Cv[n] + D*u ; y *= silu(z)
// ---------------------------------------------------------------------------
__global__ __launch_bounds__(256)
void mamba_scan(const float* __restrict__ xdbl,    // (B*L, 96) fp32
                const float* __restrict__ dt,      // (B*L, DIN) fp32
                const float* __restrict__ xc,      // (B*L, DIN) fp32
                const __bf16* __restrict__ xz,     // (B*L, 2*DIN) bf16 (z gate)
                const float* __restrict__ b_dt,
                const float* __restrict__ A_log,   // (DIN, DST)
                const float* __restrict__ Dvec,
                __bf16* __restrict__ y_bf)         // (B*L, DIN) bf16 out
{
    const int gid = blockIdx.x * blockDim.x + threadIdx.x;
    const int n   = gid & (DST - 1);
    const int dgl = gid >> 4;
    const int b   = dgl / DIN;
    const int d   = dgl % DIN;

    const float a_coef = -__expf(A_log[d * DST + n]);
    const float bias2  = 2.0f * b_dt[d];
    const float Dd     = Dvec[d];

    float h = 0.0f;
    const size_t row0 = (size_t)b * LSEQ;

    for (int l = 0; l < LSEQ; ++l) {
        const size_t row = row0 + l;
        const float Bv = xdbl[row * XPROJ_N + RDT + n];
        const float Cv = xdbl[row * XPROJ_N + RDT + DST + n];
        const float u  = xc[row * DIN + d];
        const float dv = dt[row * DIN + d] + bias2;
        const float delta = (dv > 20.0f) ? dv : log1pf(__expf(dv));

        h = fmaf(__expf(delta * a_coef), h, delta * Bv * u);
        float part = h * Cv;
        #pragma unroll
        for (int off = 8; off >= 1; off >>= 1)
            part += __shfl_xor(part, off, 16);

        if (n == 0) {
            float y = part + Dd * u;
            const float z = (float)xz[row * (2 * DIN) + DIN + d];
            y *= z / (1.0f + __expf(-z));
            y_bf[row * DIN + d] = (__bf16)y;
        }
    }
}

// ---------------------------------------------------------------------------
// Launcher. Inputs: x, W_in, W_conv, b_conv, W_xproj, W_dt, b_dt, A_log, D,
// W_out. Workspace layout (bf16 region then fp32 region), ~151 MB total.
// ---------------------------------------------------------------------------
extern "C" void kernel_launch(void* const* d_in, const int* in_sizes, int n_in,
                              void* d_out, int out_size, void* d_ws, size_t ws_size,
                              hipStream_t stream)
{
    const float* x       = (const float*)d_in[0];
    const float* W_in    = (const float*)d_in[1];
    const float* W_conv  = (const float*)d_in[2];
    const float* b_conv  = (const float*)d_in[3];
    const float* W_xproj = (const float*)d_in[4];
    const float* W_dt    = (const float*)d_in[5];
    const float* b_dt    = (const float*)d_in[6];
    const float* A_log   = (const float*)d_in[7];
    const float* Dvec    = (const float*)d_in[8];
    const float* W_out   = (const float*)d_in[9];
    float* out = (float*)d_out;

    __bf16* x_bf    = (__bf16*)d_ws;                          // 4096*1024
    __bf16* Win_bf  = x_bf    + (size_t)NROW * DM;            // 4096*1024
    __bf16* xz_bf   = Win_bf  + (size_t)(2 * DIN) * DM;       // 4096*4096
    __bf16* xc_bf   = xz_bf   + (size_t)NROW * (2 * DIN);     // 4096*2048
    __bf16* Wxp_bf  = xc_bf   + (size_t)NROW * DIN;           // 128*2048
    __bf16* dtlo_bf = Wxp_bf  + (size_t)XPROJ_PAD * DIN;      // 4096*64
    __bf16* Wdt_bf  = dtlo_bf + (size_t)NROW * RDT;           // 2048*64
    __bf16* y_bf    = Wdt_bf  + (size_t)DIN * RDT;            // 4096*2048
    __bf16* Wout_bf = y_bf    + (size_t)NROW * DIN;           // 1024*2048
    float*  xc      = (float*)(Wout_bf + (size_t)DM * DIN);   // 4096*2048
    float*  xdbl    = xc   + (size_t)NROW * DIN;              // 4096*96
    float*  dt      = xdbl + (size_t)NROW * XPROJ_N;          // 4096*2048

    const dim3 blk(256);

    // --- one-time bf16 conversions (weights + input activations) ---
    cvt_f32_bf16<<<(NROW * DM) / 256, blk, 0, stream>>>(x, x_bf);
    cvt_f32_bf16<<<((2 * DIN) * DM) / 256, blk, 0, stream>>>(W_in, Win_bf);
    cvt_pad_bf16<<<(XPROJ_PAD * DIN) / 256, blk, 0, stream>>>(W_xproj, Wxp_bf,
                                                              XPROJ_N, DIN);
    cvt_f32_bf16<<<(DIN * RDT) / 256, blk, 0, stream>>>(W_dt, Wdt_bf);
    cvt_f32_bf16<<<(DM * DIN) / 256, blk, 0, stream>>>(W_out, Wout_bf);

    // 1) xz = x @ W_in^T -> bf16 (4096,4096)
    mamba_wmma_gemm<<<dim3(NROW / 64, (2 * DIN) / 128), blk, 0, stream>>>(
        x_bf, Win_bf, xz_bf, 2 * DIN, DM, DM, 2 * DIN, /*bf16*/1);

    // 2) depthwise conv + SiLU -> xc (fp32) and xc_bf
    mamba_conv_silu<<<(NROW * DIN) / 256, blk, 0, stream>>>(
        xz_bf, W_conv, b_conv, xc, xc_bf);

    // 3) x_dbl = xc @ W_xproj^T -> fp32 (4096,96); W padded to 128 rows
    mamba_wmma_gemm<<<dim3(NROW / 64, 1), blk, 0, stream>>>(
        xc_bf, Wxp_bf, xdbl, XPROJ_N, DIN, DIN, XPROJ_N, /*f32*/0);

    // 3b) dt_lo slice -> contiguous bf16 (4096,64)
    extract_dtlo<<<(NROW * RDT) / 256, blk, 0, stream>>>(xdbl, dtlo_bf);

    // 4) dt = dt_lo @ W_dt^T -> fp32 (4096,2048)
    mamba_wmma_gemm<<<dim3(NROW / 64, DIN / 128), blk, 0, stream>>>(
        dtlo_bf, Wdt_bf, dt, DIN, RDT, RDT, DIN, /*f32*/0);

    // 5) selective scan -> y_bf, gated by silu(z)
    mamba_scan<<<(BSZ * DIN * DST) / 256, blk, 0, stream>>>(
        xdbl, dt, xc, xz_bf, b_dt, A_log, Dvec, y_bf);

    // 6) out = y @ W_out^T -> fp32 (4096,1024)
    mamba_wmma_gemm<<<dim3(NROW / 64, DM / 128), blk, 0, stream>>>(
        y_bf, Wout_bf, out, DM, DIN, DIN, DM, /*f32*/0);
}
